// MultiHeadAttention_72662256714107
// MI455X (gfx1250) — matrix operand order repaired and checked by reading, not compiled, not run on hardware
//
#include <hip/hip_runtime.h>

#define DMODEL 1024
#define NHEADS 16
#define DK     64
#define BATCH  2
#define SEQ    2048
#define MROWS  (BATCH * SEQ)   // 4096

typedef __bf16 bf16;
typedef __attribute__((ext_vector_type(16))) __bf16 v16bf;
typedef __attribute__((ext_vector_type(8)))  __bf16 v8bf;
typedef __attribute__((ext_vector_type(4)))  __bf16 v4bf;
typedef __attribute__((ext_vector_type(8)))  float  v8f;

#define NEG_INF (-__builtin_inff())

static __device__ __forceinline__ v16bf frag_join(v8bf lo, v8bf hi) {
  v16bf r;
#pragma unroll
  for (int i = 0; i < 8; ++i) { r[i] = lo[i]; r[i + 8] = hi[i]; }
  return r;
}

// A-matrix fragment (16x32 bf16, MxK): lane r = row; element j -> K = 16*half + j
// (contiguous per lane).  p points at (row0, k0); ld = row stride.
static __device__ __forceinline__ v16bf load_a_frag(const bf16* __restrict__ p, int ld,
                                                    int r, int half) {
  const bf16* base = p + (size_t)r * ld;
  v8bf lo = *(const v8bf*)(base + half * 16);
  v8bf hi = *(const v8bf*)(base + half * 16 + 8);
  return frag_join(lo, hi);
}

// B-matrix fragment (32x16 bf16, KxN) from a transposed (NxK row-major) view:
// lane r = column N; element j -> K = 16*half + j (contiguous per lane).
static __device__ __forceinline__ v16bf load_bt_frag(const bf16* __restrict__ p, int ld,
                                                     int r, int half) {
  const bf16* base = p + (size_t)r * ld + half * 16;
  v8bf lo = *(const v8bf*)(base);
  v8bf hi = *(const v8bf*)(base + 8);
  return frag_join(lo, hi);
}

static __device__ __forceinline__ v8f wmma_bf16(v16bf a, v16bf b, v8f c) {
  return __builtin_amdgcn_wmma_f32_16x16x32_bf16(false, a, false, b, (short)0, c,
                                                 false, false);
}

// Async global -> LDS copy of 16 bytes per lane (ASYNCcnt-tracked).
static __device__ __forceinline__ void async_copy_b128(unsigned lds_off,
                                                       const bf16* gptr) {
  asm volatile("global_load_async_to_lds_b128 %0, %1, off"
               :: "v"(lds_off), "v"(gptr) : "memory");
}
static __device__ __forceinline__ void wait_asynccnt0() {
  asm volatile("s_wait_asynccnt 0x0" ::: "memory");
}

// ---------------------------------------------------------------------------
// fp32 -> bf16 conversion (vectorized x4)
// ---------------------------------------------------------------------------
__global__ __launch_bounds__(256) void cvt_bf16_kernel(const float* __restrict__ in,
                                                       bf16* __restrict__ out, int n) {
  int idx = (blockIdx.x * 256 + threadIdx.x) * 4;
  if (idx >= n) return;
  float4 f = *(const float4*)(in + idx);
  v4bf o;
  o[0] = (bf16)f.x; o[1] = (bf16)f.y; o[2] = (bf16)f.z; o[3] = (bf16)f.w;
  *(v4bf*)(out + idx) = o;
}

// ---------------------------------------------------------------------------
// GEMM: Y(M x 1024) = A(M x 1024, bf16) * W(1024 x 1024, bf16)
// Block tile 256x64, 8 waves, each wave a 32x64 tile (2 M-subtiles x 4 N-subtiles,
// 8 WMMAs per 32-wide K-chunk). Per K-chunk:
//   - A block (256x32) staged global->LDS with global_load_async_to_lds_b128
//   - W chunk (32x64) staged transposed through LDS (for contiguous B-fragments)
// mode 0: bf16 row-major out; mode 1: bf16 out transposed to (b,h,d,s);
// mode 2: fp32 row-major out.
// ---------------------------------------------------------------------------
__global__ __launch_bounds__(256) void gemm_bf16_wmma(const bf16* __restrict__ A,
                                                      const bf16* __restrict__ W,
                                                      void* __restrict__ Yout, int mode) {
  __shared__ bf16 Alds[256 * 32];  // 16 KiB: A block chunk, row-major (row, k)
  __shared__ bf16 Wt[64 * 32];     //  4 KiB: Wt[n][k] for current K-chunk

  const int lane = threadIdx.x & 31;
  const int wave = threadIdx.x >> 5;
  const int r    = lane & 15;
  const int half = lane >> 4;
  const int n0   = blockIdx.x * 64;
  const int mblk = blockIdx.y * 256;
  const int mrow = wave * 32;              // wave's row offset inside the block tile

  v8f acc[2][4];
#pragma unroll
  for (int mi = 0; mi < 2; ++mi)
#pragma unroll
    for (int nd = 0; nd < 4; ++nd)
#pragma unroll
      for (int i = 0; i < 8; ++i) acc[mi][nd][i] = 0.0f;

  const int krow = threadIdx.x >> 3;        // 0..31
  const int ncol = (threadIdx.x & 7) * 8;   // 0..56
  const unsigned abase = (unsigned)(size_t)&Alds[0];

  for (int kc = 0; kc < DMODEL; kc += 32) {
    // ---- stage A block (256 rows x 32 k) via async copy: 4 x 16B per thread ----
#pragma unroll
    for (int c = 0; c < 4; ++c) {
      int chunk = threadIdx.x + c * 256;    // 0..1023
      int arow  = chunk >> 2;               // 0..255
      int apart = (chunk & 3) * 8;          // halves 0,8,16,24
      async_copy_b128(abase + (unsigned)(arow * 32 + apart) * 2,
                      A + (size_t)(mblk + arow) * DMODEL + kc + apart);
    }
    // ---- stage W chunk (32 x 64) transposed into LDS ---------------------------
    v8bf w8 = *(const v8bf*)(W + (size_t)(kc + krow) * DMODEL + n0 + ncol);
    if (kc + 32 < DMODEL)
      __builtin_prefetch(W + (size_t)(kc + 32 + krow) * DMODEL + n0 + ncol, 0, 0);
#pragma unroll
    for (int j = 0; j < 8; ++j) Wt[(ncol + j) * 32 + krow] = w8[j];

    wait_asynccnt0();
    __syncthreads();

    // ---- fragments (batched), then 8 independent WMMAs -------------------------
    v16bf a0 = load_a_frag(&Alds[(mrow)      * 32], 32, r, half);
    v16bf a1 = load_a_frag(&Alds[(mrow + 16) * 32], 32, r, half);
    v16bf b0 = load_bt_frag(&Wt[0 * 16 * 32], 32, r, half);
    v16bf b1 = load_bt_frag(&Wt[1 * 16 * 32], 32, r, half);
    v16bf b2 = load_bt_frag(&Wt[2 * 16 * 32], 32, r, half);
    v16bf b3 = load_bt_frag(&Wt[3 * 16 * 32], 32, r, half);

    acc[0][0] = wmma_bf16(a0, b0, acc[0][0]);
    acc[0][1] = wmma_bf16(a0, b1, acc[0][1]);
    acc[0][2] = wmma_bf16(a0, b2, acc[0][2]);
    acc[0][3] = wmma_bf16(a0, b3, acc[0][3]);
    acc[1][0] = wmma_bf16(a1, b0, acc[1][0]);
    acc[1][1] = wmma_bf16(a1, b1, acc[1][1]);
    acc[1][2] = wmma_bf16(a1, b2, acc[1][2]);
    acc[1][3] = wmma_bf16(a1, b3, acc[1][3]);

    __syncthreads();
  }

  // Epilogue. C element (mi, nd, i): row = mblk + mrow + mi*16 + 8*half + i,
  //                                  col = n0 + nd*16 + r.
  if (mode == 2) {
    float* Y = (float*)Yout;
#pragma unroll
    for (int mi = 0; mi < 2; ++mi)
#pragma unroll
      for (int nd = 0; nd < 4; ++nd) {
        int col = n0 + nd * 16 + r;
#pragma unroll
        for (int i = 0; i < 8; ++i) {
          int row = mblk + mrow + mi * 16 + half * 8 + i;
          Y[(size_t)row * DMODEL + col] = acc[mi][nd][i];
        }
      }
  } else if (mode == 0) {
    bf16* Y = (bf16*)Yout;
#pragma unroll
    for (int mi = 0; mi < 2; ++mi)
#pragma unroll
      for (int nd = 0; nd < 4; ++nd) {
        int col = n0 + nd * 16 + r;
#pragma unroll
        for (int i = 0; i < 8; ++i) {
          int row = mblk + mrow + mi * 16 + half * 8 + i;
          Y[(size_t)row * DMODEL + col] = (bf16)acc[mi][nd][i];
        }
      }
  } else {  // mode 1: V transposed -> Vt[(b*H + h)*64 + d][s], stride SEQ
    bf16* Y = (bf16*)Yout;
#pragma unroll
    for (int mi = 0; mi < 2; ++mi)
#pragma unroll
      for (int nd = 0; nd < 4; ++nd) {
        int col = n0 + nd * 16 + r;
        int hh  = col >> 6;
        int d   = col & 63;
#pragma unroll
        for (int i = 0; i < 8; ++i) {
          int row = mblk + mrow + mi * 16 + half * 8 + i;
          int b   = row >> 11;       // row / SEQ
          int s   = row & 2047;      // row % SEQ
          Y[((size_t)(b * NHEADS + hh) * DK + d) * SEQ + s] = (bf16)acc[mi][nd][i];
        }
      }
  }
}

// ---------------------------------------------------------------------------
// Flash attention, causal. One wave per block handles a 16-row q-tile of one
// (batch, head). Keys processed 32 at a time: 4 WMMAs for Q*K^T, online
// softmax (row reductions via shfl_xor within 16-lane half-groups), P goes
// through LDS (C-layout -> A-layout transpose), 4 WMMAs for P*V.
// ---------------------------------------------------------------------------
__global__ __launch_bounds__(32) void attn_wmma(const bf16* __restrict__ Q,
                                                const bf16* __restrict__ K,
                                                const bf16* __restrict__ Vt,
                                                bf16* __restrict__ O) {
  __shared__ bf16 Plds[16 * 32];

  const int lane = threadIdx.x & 31;
  const int r    = lane & 15;
  const int half = lane >> 4;
  const int q0   = blockIdx.x * 16;
  const int h    = blockIdx.y;
  const int b    = blockIdx.z;

  const size_t qk_base = (size_t)(b * SEQ) * DMODEL + h * DK;
  const bf16*  vbase   = Vt + (size_t)(b * NHEADS + h) * DK * SEQ;  // row stride SEQ

  // Q fragments for the whole d_k = 64 (two 32-wide K-chunks)
  v16bf qa0 = load_a_frag(Q + qk_base + (size_t)q0 * DMODEL + 0,  DMODEL, r, half);
  v16bf qa1 = load_a_frag(Q + qk_base + (size_t)q0 * DMODEL + 32, DMODEL, r, half);

  float m[8], l[8];
  v8f   o[4];
#pragma unroll
  for (int i = 0; i < 8; ++i) { m[i] = NEG_INF; l[i] = 0.0f; }
#pragma unroll
  for (int nd = 0; nd < 4; ++nd)
#pragma unroll
    for (int i = 0; i < 8; ++i) o[nd][i] = 0.0f;

  const float scale = 0.125f;  // 1/sqrt(64)

  for (int k0 = 0; k0 < q0 + 16; k0 += 32) {
    // ---- scores: s0 = keys k0..k0+15, s1 = keys k0+16..k0+31 -----------------
    v8f s0, s1;
#pragma unroll
    for (int i = 0; i < 8; ++i) { s0[i] = 0.0f; s1[i] = 0.0f; }
    {
      const bf16* kb = K + qk_base + (size_t)k0 * DMODEL;
      v16bf k00 = load_bt_frag(kb + 0,  DMODEL, r, half);
      v16bf k01 = load_bt_frag(kb + 32, DMODEL, r, half);
      v16bf k10 = load_bt_frag(kb + (size_t)16 * DMODEL + 0,  DMODEL, r, half);
      v16bf k11 = load_bt_frag(kb + (size_t)16 * DMODEL + 32, DMODEL, r, half);
      s0 = wmma_bf16(qa0, k00, s0);
      s0 = wmma_bf16(qa1, k01, s0);
      s1 = wmma_bf16(qa0, k10, s1);
      s1 = wmma_bf16(qa1, k11, s1);
    }

    // ---- online softmax update ----------------------------------------------
#pragma unroll
    for (int i = 0; i < 8; ++i) {
      int   qrow = q0 + half * 8 + i;
      float v0 = s0[i] * scale;
      float v1 = s1[i] * scale;
      if (k0 + r      > qrow) v0 = NEG_INF;
      if (k0 + 16 + r > qrow) v1 = NEG_INF;

      float mx = fmaxf(v0, v1);
      mx = fmaxf(mx, __shfl_xor(mx, 1));
      mx = fmaxf(mx, __shfl_xor(mx, 2));
      mx = fmaxf(mx, __shfl_xor(mx, 4));
      mx = fmaxf(mx, __shfl_xor(mx, 8));

      float mnew  = fmaxf(m[i], mx);
      float alpha = __expf(m[i] - mnew);
      float p0    = __expf(v0 - mnew);
      float p1    = __expf(v1 - mnew);

      float rs = p0 + p1;
      rs += __shfl_xor(rs, 1);
      rs += __shfl_xor(rs, 2);
      rs += __shfl_xor(rs, 4);
      rs += __shfl_xor(rs, 8);

      l[i] = l[i] * alpha + rs;
      m[i] = mnew;
#pragma unroll
      for (int nd = 0; nd < 4; ++nd) o[nd][i] *= alpha;

      // P (C-layout) -> LDS row-major 16x32
      int prow = half * 8 + i;
      Plds[prow * 32 + r]      = (bf16)p0;
      Plds[prow * 32 + 16 + r] = (bf16)p1;
    }

    // ---- P * V ---------------------------------------------------------------
    // LDS ops are in-order within a wave: loads below see the stores above.
    v16bf pf = load_a_frag(Plds, 32, r, half);
    v16bf vf0 = load_bt_frag(vbase + (size_t)(0 * 16) * SEQ + k0, SEQ, r, half);
    v16bf vf1 = load_bt_frag(vbase + (size_t)(1 * 16) * SEQ + k0, SEQ, r, half);
    v16bf vf2 = load_bt_frag(vbase + (size_t)(2 * 16) * SEQ + k0, SEQ, r, half);
    v16bf vf3 = load_bt_frag(vbase + (size_t)(3 * 16) * SEQ + k0, SEQ, r, half);
    o[0] = wmma_bf16(pf, vf0, o[0]);
    o[1] = wmma_bf16(pf, vf1, o[1]);
    o[2] = wmma_bf16(pf, vf2, o[2]);
    o[3] = wmma_bf16(pf, vf3, o[3]);
  }

  // ---- epilogue: normalize and store (b, s, h*64+d) bf16 ---------------------
#pragma unroll
  for (int i = 0; i < 8; ++i) {
    int   qrow = q0 + half * 8 + i;
    float inv  = 1.0f / l[i];
#pragma unroll
    for (int nd = 0; nd < 4; ++nd) {
      O[(size_t)(b * SEQ + qrow) * DMODEL + h * DK + nd * 16 + r] =
          (bf16)(o[nd][i] * inv);
    }
  }
}

// ---------------------------------------------------------------------------
// Launch
// ---------------------------------------------------------------------------
extern "C" void kernel_launch(void* const* d_in, const int* in_sizes, int n_in,
                              void* d_out, int out_size, void* d_ws, size_t ws_size,
                              hipStream_t stream) {
  const float* x  = (const float*)d_in[0];
  const float* Wq = (const float*)d_in[1];
  const float* Wk = (const float*)d_in[2];
  const float* Wv = (const float*)d_in[3];
  const float* Wo = (const float*)d_in[4];

  char* ws = (char*)d_ws;
  const size_t MB = 1u << 20;
  bf16* xb = (bf16*)(ws + 0 * MB);    // 8 MiB  (B*S, D) bf16
  bf16* wq = (bf16*)(ws + 8 * MB);    // 2 MiB
  bf16* wk = (bf16*)(ws + 10 * MB);   // 2 MiB
  bf16* wv = (bf16*)(ws + 12 * MB);   // 2 MiB
  bf16* wo = (bf16*)(ws + 14 * MB);   // 2 MiB
  bf16* Qb = (bf16*)(ws + 16 * MB);   // 8 MiB  (B*S, D)
  bf16* Kb = (bf16*)(ws + 24 * MB);   // 8 MiB  (B*S, D)
  bf16* Vt = (bf16*)(ws + 32 * MB);   // 8 MiB  (B*H*dk, S)
  bf16* Ob = (bf16*)(ws + 40 * MB);   // 8 MiB  (B*S, D)

  const int nx = BATCH * SEQ * DMODEL;  // 4194304
  const int nw = DMODEL * DMODEL;       // 1048576
  cvt_bf16_kernel<<<nx / 1024, 256, 0, stream>>>(x,  xb, nx);
  cvt_bf16_kernel<<<nw / 1024, 256, 0, stream>>>(Wq, wq, nw);
  cvt_bf16_kernel<<<nw / 1024, 256, 0, stream>>>(Wk, wk, nw);
  cvt_bf16_kernel<<<nw / 1024, 256, 0, stream>>>(Wv, wv, nw);
  cvt_bf16_kernel<<<nw / 1024, 256, 0, stream>>>(Wo, wo, nw);

  dim3 ggrid(DMODEL / 64, MROWS / 256);
  gemm_bf16_wmma<<<ggrid, 256, 0, stream>>>(xb, wq, Qb, 0);
  gemm_bf16_wmma<<<ggrid, 256, 0, stream>>>(xb, wk, Kb, 0);
  gemm_bf16_wmma<<<ggrid, 256, 0, stream>>>(xb, wv, Vt, 1);

  dim3 agrid(SEQ / 16, NHEADS, BATCH);
  attn_wmma<<<agrid, 32, 0, stream>>>(Qb, Kb, Vt, Ob);

  gemm_bf16_wmma<<<ggrid, 256, 0, stream>>>(Ob, wo, d_out, 2);
}
